// mergeScales_137438953763
// MI455X (gfx1250) — matrix operand order repaired and checked by reading, not compiled, not run on hardware
//
#include <hip/hip_runtime.h>
#include <hip/hip_bf16.h>

// MI455X (gfx1250): conv + FC via V_WMMA_F32_16X16X32_F16 (wave32).
// NHWC activations with 1-pixel zero halo (branchless implicit-GEMM conv),
// fragment-major packed weights, 4 pixel-tile / 4 N-tile register blocking.

typedef __attribute__((ext_vector_type(16))) _Float16 v16h;
typedef __attribute__((ext_vector_type(8)))  _Float16 v8h;
typedef __attribute__((ext_vector_type(8)))  float    v8f;

#define NB 768           // 128 batch * 6 objects
#define LRELU(v) ((v) > 0.0f ? (v) : 0.01f * (v))

// --------------------------------------------------------- zero (halo init)
__global__ void zero_f16(_Float16* __restrict__ p, int total8) {
  int i = blockIdx.x * blockDim.x + threadIdx.x;
  if (i >= total8) return;
  *(v8h*)(p + (size_t)i * 8) = (v8h){};
}

// ------------------------------------- normalize -> padded NHWC [34,34,32]
__global__ void normalize_nhwc(const float* __restrict__ x, const float* __restrict__ nv,
                               _Float16* __restrict__ o, int total) {
  int i = blockIdx.x * blockDim.x + threadIdx.x;
  if (i >= total) return;                    // i = ((n*1024)+p)*32 + cp
  int cp = i & 31;
  int p  = (i >> 5) & 1023;
  int n  = i >> 15;
  int h = p >> 5, w = p & 31;
  float v = 0.0f;
  if (cp < 10) {
    float xx = x[((size_t)n * 10 + cp) * 1024 + p];
    float m = nv[cp * 2 + 0], s = nv[cp * 2 + 1];
    v = (xx != 0.0f) ? (xx - m) / s : 0.0f;
  }
  o[(((size_t)n * 34 + h + 1) * 34 + w + 1) * 32 + cp] = (_Float16)v;
}

// --------------------------- conv weights OIHW f32 -> fragment-major f16
// K-space: k = tap*Cpad + ci (tap = kh*3+kw), Kpad = 9*Cpad.
__global__ void pack_convw(const float* __restrict__ w, _Float16* __restrict__ o,
                           int Cout, int Cin, int Cpad) {
  int Kpad = 9 * Cpad;
  int i = blockIdx.x * blockDim.x + threadIdx.x;
  if (i >= Cout * Kpad) return;
  int co = i / Kpad, k = i - co * Kpad;
  int tap = k / Cpad, ci = k - tap * Cpad;
  int kh = tap / 3, kw = tap - kh * 3;
  float v = (ci < Cin) ? w[(((size_t)co * Cin + ci) * 3 + kh) * 3 + kw] : 0.0f;
  int chunk = k >> 5, wk = k & 31, g = wk >> 3;
  int slot = wk + 0 * g;
  o[(((size_t)co * (Kpad >> 5) + chunk) << 5) + slot] = (_Float16)v;
}

// ------------------- GEMM B f32 [Ksrc,N] -> fragment-major f16 [Kpad/32][N][32]
__global__ void pack_mat(const float* __restrict__ w, _Float16* __restrict__ o,
                         int Ksrc, int Kpad, int N) {
  int i = blockIdx.x * blockDim.x + threadIdx.x;
  if (i >= Kpad * N) return;
  int k = i / N, n = i - k * N;
  float v = (k < Ksrc) ? w[(size_t)k * N + n] : 0.0f;
  o[(((size_t)(k >> 5) * N + n) << 5) + (k & 31)] = (_Float16)v;
}

// FC1 weights with NCHW->NHWC flatten permutation: dst row (h*8+w)*64+c <- src row c*64+(h*8+w)
__global__ void pack_fc1(const float* __restrict__ w, _Float16* __restrict__ o) {
  int i = blockIdx.x * blockDim.x + threadIdx.x;     // 4096*512
  if (i >= 4096 * 512) return;
  int k = i / 512, n = i - k * 512;
  int hw = k >> 6, c = k & 63;
  int ks = c * 64 + hw;
  o[(((size_t)(k >> 5) * 512 + n) << 5) + (k & 31)] =
      (_Float16)w[(size_t)ks * 512 + n];
}

// ------------------------------------------------- conv3x3 as implicit GEMM
// X padded NHWC [NB,W+2,W+2,CPAD]; Y padded NHWC [NB,W+2,W+2,Cout].
// One wave = 4 pixel-tiles x 16 couts, branchless. grid (W*W/64, NB, Cout/16).
template <int CPAD, int W>
__global__ void conv3x3_wmma(const _Float16* __restrict__ X, const _Float16* __restrict__ Wp,
                             const float* __restrict__ gamma, const float* __restrict__ beta,
                             _Float16* __restrict__ Y, int Cout, int relu) {
  constexpr int LOGW = (W == 32) ? 5 : 4;
  constexpr int PW   = W + 2;
  constexpr int NCH  = (9 * CPAD) >> 5;      // k-chunks total
  const int lane = threadIdx.x;
  const int n    = blockIdx.y;
  const int mt   = blockIdx.z;
  const int pix0 = blockIdx.x * 64;
  const _Float16* Xn = X + (size_t)n * PW * PW * CPAD;

  const int hi = lane >> 4;
  const int m  = mt * 16 + (lane & 15);
  const _Float16* wrow = Wp + (((size_t)m * NCH) << 5) + (hi << 4);

  int pc[4], ohv[4], owv[4];
  const _Float16* ctr[4];
#pragma unroll
  for (int pt = 0; pt < 4; ++pt) {
    pc[pt]  = pix0 + (lane & 15) + 16 * pt;
    ohv[pt] = pc[pt] >> LOGW;
    owv[pt] = pc[pt] & (W - 1);
    ctr[pt] = Xn + ((size_t)(ohv[pt] + 1) * PW + owv[pt] + 1) * CPAD + (hi << 4);
  }

  v8f acc[4];
#pragma unroll
  for (int pt = 0; pt < 4; ++pt) acc[pt] = (v8f){};

  int chunk = 0;
#pragma unroll
  for (int tap = 0; tap < 9; ++tap) {
    const int toff = ((tap / 3 - 1) * PW + (tap % 3 - 1)) * CPAD;  // constant after unroll
#pragma unroll
    for (int cb = 0; cb < CPAD; cb += 32, ++chunk) {
      v16h a = *(const v16h*)(wrow + ((size_t)chunk << 5));
#pragma unroll
      for (int pt = 0; pt < 4; ++pt) {
        v16h b = *(const v16h*)(ctr[pt] + toff + cb);
        acc[pt] = __builtin_amdgcn_wmma_f32_16x16x32_f16(false, a, false, b, (short)0,
                                                         acc[pt], false, false);
      }
    }
  }
  // BN + optional leaky relu; padded NHWC store: 8 contiguous halves per lane
#pragma unroll
  for (int pt = 0; pt < 4; ++pt) {
    v8h out;
#pragma unroll
    for (int r = 0; r < 8; ++r) {
      int mm = mt * 16 + (hi << 3) + r;
      float sc = gamma[mm] * rsqrtf(1.0f + 1e-5f);
      float v  = acc[pt][r] * sc + beta[mm];
      if (relu) v = LRELU(v);
      out[r] = (_Float16)v;
    }
    *(v8h*)(Y + (((size_t)n * PW + ohv[pt] + 1) * PW + owv[pt] + 1) * Cout +
            mt * 16 + (hi << 3)) = out;
  }
}

// --------------------------------------------------------------- WMMA GEMM
// O = act(A[M,lda]f16 * Bpacked + bias); one wave = NT adjacent N-tiles.
template <int NT>
__global__ void gemm_wmma(const _Float16* __restrict__ A, int lda,
                          const _Float16* __restrict__ B, int N,
                          const float* __restrict__ bias,
                          _Float16* __restrict__ Oh, float* __restrict__ Of,
                          int ldo, int col0, int Kpad, int relu) {
  const int lane = threadIdx.x;
  const int hi = lane >> 4;
  const int m  = blockIdx.y * 16 + (lane & 15);
  const int nc0 = blockIdx.x * 16 * NT + (lane & 15);
  const _Float16* arow = A + (size_t)m * lda;

  v8f acc[NT];
#pragma unroll
  for (int t = 0; t < NT; ++t) acc[t] = (v8f){};

  for (int kb = 0; kb < Kpad; kb += 32) {
    const _Float16* ap = arow + kb + (hi << 4);
    v8h lo = *(const v8h*)ap;
    v8h hh = *(const v8h*)(ap + 8);
    v16h a;
#pragma unroll
    for (int e = 0; e < 8; ++e) { a[e] = lo[e]; a[8 + e] = hh[e]; }
    const _Float16* bbase = B + (((size_t)(kb >> 5) * N + nc0) << 5) + (hi << 4);
#pragma unroll
    for (int t = 0; t < NT; ++t) {
      v16h b = *(const v16h*)(bbase + ((size_t)t << 9));   // +16 cols = 16*32 halves
      acc[t] = __builtin_amdgcn_wmma_f32_16x16x32_f16(false, a, false, b, (short)0,
                                                      acc[t], false, false);
    }
  }
#pragma unroll
  for (int t = 0; t < NT; ++t) {
    int nc = nc0 + 16 * t;
#pragma unroll
    for (int r = 0; r < 8; ++r) {
      int mm = blockIdx.y * 16 + r + 8 * hi;
      float v = acc[t][r] + (bias ? bias[nc] : 0.0f);
      if (relu) v = LRELU(v);
      if (Oh) Oh[(size_t)mm * ldo + col0 + nc] = (_Float16)v;
      if (Of) Of[(size_t)mm * ldo + col0 + nc] = v;
    }
  }
}

// ------------------------------------------------------- channel attention
// U padded NHWC [NB,W+2,W+2,C]: s = GAP; e = sigmoid(lrelu(s@W1^T)@W2^T).
__global__ void chan_attn(const _Float16* __restrict__ U, const float* __restrict__ W1,
                          const float* __restrict__ W2, float* __restrict__ E,
                          int C, int Wd) {
  __shared__ float s[64];
  __shared__ float hid[8];
  int n = blockIdx.x, c = threadIdx.x;
  int R = C / 8, PW = Wd + 2;
  const _Float16* u = U + (size_t)n * PW * PW * C + c;
  float acc = 0.0f;
  for (int h = 0; h < Wd; ++h)
    for (int w = 0; w < Wd; ++w)
      acc += (float)u[(size_t)((h + 1) * PW + w + 1) * C];
  s[c] = acc / (float)(Wd * Wd);
  __syncthreads();
  if (c < R) {
    float h = 0.0f;
    for (int k = 0; k < C; ++k) h += s[k] * W1[c * C + k];
    hid[c] = LRELU(h);
  }
  __syncthreads();
  float ex = 0.0f;
  for (int j = 0; j < R; ++j) ex += hid[j] * W2[c * R + j];
  E[(size_t)n * C + c] = 1.0f / (1.0f + expf(-ex));
}

// -------- resblock epilogue + lrelu + 2x2 pool; padded in, padded/dense out
__global__ void res_pool(const _Float16* __restrict__ U, const _Float16* __restrict__ S,
                         const float* __restrict__ E, _Float16* __restrict__ P,
                         int C, int Wd, int opad, int total) {
  int i = blockIdx.x * blockDim.x + threadIdx.x;
  if (i >= total) return;
  int PW = Wd + 2, W2 = Wd >> 1, OPW = W2 + 2;
  int c = i % C; int t = i / C;
  int ow = t % W2; t /= W2;
  int oh = t % W2; int n = t / W2;
  float e = E[(size_t)n * C + c];
  float acc = 0.0f;
#pragma unroll
  for (int dy = 0; dy < 2; ++dy)
#pragma unroll
    for (int dx = 0; dx < 2; ++dx) {
      size_t idx = (((size_t)n * PW + 2 * oh + dy + 1) * PW + 2 * ow + dx + 1) * C + c;
      float v = (float)S[idx] + (float)U[idx] * e;
      acc += LRELU(v);
    }
  float r = acc * 0.25f;
  if (opad) P[(((size_t)n * OPW + oh + 1) * OPW + ow + 1) * C + c] = (_Float16)r;
  else      P[i] = (_Float16)r;
}

// ------------------------- scale attention: BN1d+tanh, softmax, scale, concat
__global__ void scale_attn(const float* __restrict__ lin, const _Float16* __restrict__ fus,
                           const float* __restrict__ g, const float* __restrict__ bb,
                           const float* __restrict__ x4, _Float16* __restrict__ zp) {
  __shared__ float red[448];
  int n = blockIdx.x, t = threadIdx.x;
  float h = 0.0f;
  if (t < 384) {
    float s = g[t] * rsqrtf(1.0f + 1e-5f);
    h = tanhf(lin[(size_t)n * 384 + t] * s + bb[t]);
  }
  red[t] = (t < 384) ? h : -3.4e38f;
  __syncthreads();
  for (int s2 = 256; s2 > 0; s2 >>= 1) {
    if (t < s2 && t + s2 < 448) red[t] = fmaxf(red[t], red[t + s2]);
    __syncthreads();
  }
  float mx = red[0];
  __syncthreads();
  float ex = (t < 384) ? expf(h - mx) : 0.0f;
  red[t] = ex;
  __syncthreads();
  for (int s2 = 256; s2 > 0; s2 >>= 1) {
    if (t < s2 && t + s2 < 448) red[t] += red[t + s2];
    __syncthreads();
  }
  float sum = red[0];
  _Float16 out;
  if (t < 384)       out = (_Float16)((float)fus[(size_t)n * 384 + t] * (ex / sum));
  else if (t < 420)  out = (_Float16)x4[(size_t)(n / 6) * 36 + (t - 384)];
  else               out = (_Float16)0.0f;
  zp[(size_t)n * 448 + t] = out;
}

// ----------------------------------------------------------- final 32 -> 7
__global__ void head7(const _Float16* __restrict__ h, const float* __restrict__ w2,
                      const float* __restrict__ b2, float* __restrict__ out) {
  int n = blockIdx.x, t = threadIdx.x;
  if (t >= 7) return;
  float acc = b2[t];
  for (int k = 0; k < 32; ++k) acc += (float)h[(size_t)n * 32 + k] * w2[k * 7 + t];
  out[(size_t)n * 7 + t] = acc;
}

// ===========================================================================
extern "C" void kernel_launch(void* const* d_in, const int* in_sizes, int n_in,
                              void* d_out, int out_size, void* d_ws, size_t ws_size,
                              hipStream_t stream) {
  (void)in_sizes; (void)n_in; (void)out_size; (void)ws_size;
  auto F = [&](int i) -> const float* { return (const float*)d_in[i]; };

  char* ws = (char*)d_ws; size_t off = 0;
  auto alloc = [&](size_t bytes) -> void* {
    void* p = ws + off; off += (bytes + 255) & ~(size_t)255; return p;
  };
  const size_t SZ34 = (size_t)NB * 34 * 34 * 32;   // padded 32x32, 32ch
  const size_t SZ18a = (size_t)NB * 18 * 18 * 32;  // padded 16x16, 32ch
  const size_t SZ18b = (size_t)NB * 18 * 18 * 64;  // padded 16x16, 64ch
  _Float16* XN = (_Float16*)alloc(SZ34 * 2);       // normalized input
  _Float16* A1 = (_Float16*)alloc(SZ34 * 2);       // conv1 out
  _Float16* T1 = (_Float16*)alloc(SZ34 * 2);       // rb1 conv1 out
  _Float16* U1 = (_Float16*)alloc(SZ34 * 2);       // rb1 conv2 out
  _Float16* P1 = (_Float16*)alloc(SZ18a * 2);      // pooled 1
  _Float16* T2 = (_Float16*)alloc(SZ18b * 2);      // rb2 conv1 out
  _Float16* U2 = (_Float16*)alloc(SZ18b * 2);      // rb2 conv2 out
  _Float16* S2 = (_Float16*)alloc(SZ18b * 2);      // rb2 shortcut out
  _Float16* P2  = (_Float16*)alloc((size_t)NB * 4096 * 2);  // dense [NB,8,8,64]
  _Float16* F1  = (_Float16*)alloc((size_t)NB * 512 * 2);
  _Float16* FUS = (_Float16*)alloc((size_t)NB * 384 * 2);
  float*    LIN = (float*)   alloc((size_t)NB * 384 * 4);
  _Float16* ZP  = (_Float16*)alloc((size_t)NB * 448 * 2);
  _Float16* H32 = (_Float16*)alloc((size_t)NB * 32 * 2);
  float*    Eb  = (float*)   alloc((size_t)NB * 64 * 4);
  _Float16* WPC = (_Float16*)alloc((size_t)64 * 576 * 2);
  _Float16* WPM = (_Float16*)alloc((size_t)4096 * 512 * 2);

  auto zero = [&](_Float16* p, size_t halves) {
    int t8 = (int)(halves / 8);
    zero_f16<<<(t8 + 255) / 256, 256, 0, stream>>>(p, t8);
  };
  // zero halo-carrying conv-input buffers once per launch
  zero(XN, SZ34); zero(A1, SZ34); zero(T1, SZ34); zero(P1, SZ18a); zero(T2, SZ18b);

  auto conv = [&](const _Float16* Xp, const float* wf, const float* gp, const float* bp,
                  _Float16* Yp, int Cin, int Cpad, int Cout, int W, int relu) {
    int Kpad = 9 * Cpad;
    pack_convw<<<(Cout * Kpad + 255) / 256, 256, 0, stream>>>(wf, WPC, Cout, Cin, Cpad);
    dim3 g(W * W / 64, NB, Cout / 16);
    if (Cpad == 32 && W == 32)
      conv3x3_wmma<32, 32><<<g, 32, 0, stream>>>(Xp, WPC, gp, bp, Yp, Cout, relu);
    else if (Cpad == 32 && W == 16)
      conv3x3_wmma<32, 16><<<g, 32, 0, stream>>>(Xp, WPC, gp, bp, Yp, Cout, relu);
    else
      conv3x3_wmma<64, 16><<<g, 32, 0, stream>>>(Xp, WPC, gp, bp, Yp, Cout, relu);
  };
  auto gemm = [&](const _Float16* A, int lda, int N, const float* bias,
                  _Float16* Oh, float* Of, int ldo, int col0, int Kpad, int relu) {
    if (N % 64 == 0) {
      dim3 g(N / 64, NB / 16);
      gemm_wmma<4><<<g, 32, 0, stream>>>(A, lda, WPM, N, bias, Oh, Of, ldo, col0, Kpad, relu);
    } else {
      dim3 g(N / 32, NB / 16);
      gemm_wmma<2><<<g, 32, 0, stream>>>(A, lda, WPM, N, bias, Oh, Of, ldo, col0, Kpad, relu);
    }
  };

  // d_in: x1,x2,x3,x4,norm_val, then per branch:
  // sc,sg,sb, rb1{c1,g1,b1,c2,g2,b2,aw1,aw2}, rb2{..+cs,gs,bs}, fc{w1,b1,w2,b2}
  const int PB[3] = {5, 5 + 26, 5 + 52};
  const float* nv = F(4);

  for (int b = 0; b < 3; ++b) {
    int p = PB[b];
    normalize_nhwc<<<(NB * 1024 * 32 + 255) / 256, 256, 0, stream>>>(
        F(b), nv, XN, NB * 1024 * 32);
    // shallow conv 10(pad32)->32 @32x32, BN+lrelu
    conv(XN, F(p + 0), F(p + 1), F(p + 2), A1, 10, 32, 32, 32, 1);
    // resblock 1 (identity shortcut)
    conv(A1, F(p + 3), F(p + 4), F(p + 5), T1, 32, 32, 32, 32, 1);
    conv(T1, F(p + 6), F(p + 7), F(p + 8), U1, 32, 32, 32, 32, 0);
    chan_attn<<<NB, 32, 0, stream>>>(U1, F(p + 9), F(p + 10), Eb, 32, 32);
    res_pool<<<(NB * 32 * 256 + 255) / 256, 256, 0, stream>>>(
        U1, A1, Eb, P1, 32, 32, 1, NB * 32 * 256);            // -> padded 18x18x32
    // resblock 2 (conv shortcut), 16x16
    conv(P1, F(p + 11), F(p + 12), F(p + 13), T2, 32, 32, 64, 16, 1);
    conv(T2, F(p + 14), F(p + 15), F(p + 16), U2, 64, 64, 64, 16, 0);
    chan_attn<<<NB, 64, 0, stream>>>(U2, F(p + 17), F(p + 18), Eb, 64, 16);
    conv(P1, F(p + 19), F(p + 20), F(p + 21), S2, 32, 32, 64, 16, 0);  // shortcut
    res_pool<<<(NB * 64 * 64 + 255) / 256, 256, 0, stream>>>(
        U2, S2, Eb, P2, 64, 16, 0, NB * 64 * 64);             // -> dense [NB,8,8,64]
    // FC 4096->512 (lrelu, perm-packed), 512->128 (lrelu) into fused columns
    pack_fc1<<<(4096 * 512 + 255) / 256, 256, 0, stream>>>(F(p + 22), WPM);
    gemm(P2, 4096, 512, F(p + 23), F1, nullptr, 512, 0, 4096, 1);
    pack_mat<<<(512 * 128 + 255) / 256, 256, 0, stream>>>(F(p + 24), WPM, 512, 512, 128);
    gemm(F1, 512, 128, F(p + 25), FUS, nullptr, 384, b * 128, 512, 1);
  }

  // scale attention: 384x384 GEMM -> BN+tanh -> softmax -> scale -> concat global
  const int PA = 5 + 78;
  pack_mat<<<(384 * 384 + 255) / 256, 256, 0, stream>>>(F(PA + 0), WPM, 384, 384, 384);
  gemm(FUS, 384, 384, F(PA + 1), nullptr, LIN, 384, 0, 384, 0);
  scale_attn<<<NB, 448, 0, stream>>>(LIN, FUS, F(PA + 2), F(PA + 3), F(3), ZP);
  // head: 420(->448 pad)->32 lrelu, then 32->7
  pack_mat<<<(448 * 32 + 255) / 256, 256, 0, stream>>>(F(PA + 4), WPM, 420, 448, 32);
  gemm(ZP, 448, 32, F(PA + 5), H32, nullptr, 32, 0, 448, 1);
  head7<<<NB, 32, 0, stream>>>(H32, F(PA + 6), F(PA + 7), (float*)d_out);
}
